// FeatureAggregationModule_54606214202019
// MI455X (gfx1250) — matrix operand order repaired and checked by reading, not compiled, not run on hardware
//
#include <hip/hip_runtime.h>

typedef unsigned short u16;
typedef unsigned char  u8;
typedef unsigned int   u32;

typedef float  v8f   __attribute__((ext_vector_type(8)));
typedef __bf16 v16bf __attribute__((ext_vector_type(16)));
typedef u32    uvec4 __attribute__((ext_vector_type(4)));

#define NROW 3072   // N tokens
#define CCH  256    // common channels
#define NHEAD 4
#define HD   64

__device__ __forceinline__ u16 f2bf(float f) {
    u32 u = __builtin_bit_cast(u32, f);
    u32 r = u + 0x7FFFu + ((u >> 16) & 1u);   // round-to-nearest-even
    return (u16)(r >> 16);
}

struct U8x32 { uvec4 lo, hi; };
__device__ __forceinline__ v16bf make_frag(uvec4 lo, uvec4 hi) {
    U8x32 u; u.lo = lo; u.hi = hi;
    return __builtin_bit_cast(v16bf, u);
}

// ---------------------------------------------------------------------------
// Generic bf16 GEMM: C[M,Nn] = A[M,K] * B, fp32 accumulate via WMMA.
//   BT=true : B is weight-like, B[n][k] (row-major [Nn,K])  -> C = A * B^T
//   BT=false: B is normal,     B[k][n] (row-major [K,Nn])   -> C = A * B
// Epilogues (EPI):
//   0: f32 out            1: bf16 out
//   2: +bias, f32 out     3: +bias, bf16 out
//   4: *alpha*keyvec[n], f32 out (per-key scaled attention logits)
//   5: *alpha, f32 out
//   6: (acc*alpha > beta) ? 1 : 0, u8 out (gram-similarity mask)
// Block: 128x128 tile, BK=64 (2 WMMA k-steps per barrier pair),
// 256 threads = 8 waves, each wave does a 32x64 (MxN) slab = 16 WMMA / k-block.
// Requires: M % 128 == 0, K % 64 == 0 (true for all launches here).
// ---------------------------------------------------------------------------
template<bool BT, int EPI>
__global__ __launch_bounds__(256) void gemm_bf16(
    const u16* __restrict__ A, int lda,
    const u16* __restrict__ Bm, int ldb,
    void* __restrict__ Cv, int ldc,
    int M, int Nn, int K,
    const float* __restrict__ bias, float alpha, float beta,
    const float* __restrict__ keyvec)
{
    __shared__ u16 As[128][72];   // [m][k0..63], row stride 144B (16B-aligned)
    __shared__ u16 Bs[128][72];   // [n][k0..63]

    const int t    = threadIdx.x;
    const int m0   = blockIdx.y * 128;
    const int n0   = blockIdx.x * 128;
    const int wave = t >> 5;
    const int lane = t & 31;
    const int half = lane >> 4;
    const int lm   = lane & 15;
    const int wm   = wave & 3;   // 4 waves along M (32 rows each)
    const int wn   = wave >> 2;  // 2 waves along N (64 cols each)

    v8f acc[2][4];
    #pragma unroll
    for (int i = 0; i < 2; i++)
        #pragma unroll
        for (int j = 0; j < 4; j++)
            acc[i][j] = (v8f)0.0f;

    for (int k0 = 0; k0 < K; k0 += 64) {
        // ---- stage A tile (128x64): 4x 16B loads per thread ----
        {
            int r = t >> 1;
            const u16* pa = A + (size_t)(m0 + r) * lda + k0;
            if (k0 + 64 < K)   // prefetch next K panel (global_prefetch_b8)
                __builtin_prefetch(pa + 64 + (t & 1) * 32, 0, 0);
            #pragma unroll
            for (int i = 0; i < 4; i++) {
                int c = (t & 1) * 32 + i * 8;
                *(uvec4*)&As[r][c] = *(const uvec4*)(pa + c);
            }
        }
        // ---- stage B tile into Bs[n][k] ----
        if (BT) {
            int n = t >> 1;
            const u16* pb = Bm + (size_t)(n0 + n) * ldb + k0;
            bool ok = (n0 + n < Nn);
            #pragma unroll
            for (int i = 0; i < 4; i++) {
                int c = (t & 1) * 32 + i * 8;
                uvec4 v = (uvec4)0u;
                if (ok) v = *(const uvec4*)(pb + c);
                *(uvec4*)&Bs[n][c] = v;
            }
        } else {
            int k = t >> 2;
            const u16* pb = Bm + (size_t)(k0 + k) * ldb + n0;
            #pragma unroll
            for (int i = 0; i < 4; i++) {
                int nc = (t & 3) * 32 + i * 8;
                uvec4 v = (uvec4)0u;
                if (n0 + nc < Nn) v = *(const uvec4*)(pb + nc);
                union { uvec4 u; u16 s[8]; } uu; uu.u = v;
                #pragma unroll
                for (int j = 0; j < 8; j++) Bs[nc + j][k] = uu.s[j];
            }
        }
        __syncthreads();

        // ---- 2 WMMA k-steps per staged block ----
        #pragma unroll
        for (int ks = 0; ks < 64; ks += 32) {
            v16bf af[2], bfr[4];
            #pragma unroll
            for (int mt = 0; mt < 2; mt++) {
                int r = wm * 32 + mt * 16 + lm;
                uvec4 lo = *(const uvec4*)&As[r][ks + half * 8];       // K 0-7 / 8-15
                uvec4 hi = *(const uvec4*)&As[r][ks + 16 + half * 8];  // K 16-23 / 24-31
                af[mt] = make_frag(lo, hi);
            }
            #pragma unroll
            for (int nt = 0; nt < 4; nt++) {
                int r = wn * 64 + nt * 16 + lm;
                uvec4 lo = *(const uvec4*)&Bs[r][ks + half * 8];       // K 0-7 / 8-15
                uvec4 hi = *(const uvec4*)&Bs[r][ks + 16 + half * 8];  // K 16-23 / 24-31
                bfr[nt] = make_frag(lo, hi);
            }
            #pragma unroll
            for (int mt = 0; mt < 2; mt++)
                #pragma unroll
                for (int nt = 0; nt < 4; nt++)
                    acc[mt][nt] = __builtin_amdgcn_wmma_f32_16x16x32_bf16(
                        false, af[mt], false, bfr[nt], (short)0, acc[mt][nt],
                        false, false);
        }
        __syncthreads();
    }

    // ---- epilogue: D layout => row M = base + half*8 + r, col N = lm ----
    #pragma unroll
    for (int mt = 0; mt < 2; mt++) {
        int mb = m0 + wm * 32 + mt * 16 + half * 8;
        #pragma unroll
        for (int nt = 0; nt < 4; nt++) {
            int n = n0 + wn * 64 + nt * 16 + lm;
            if (n >= Nn) continue;
            #pragma unroll
            for (int r = 0; r < 8; r++) {
                float v = acc[mt][nt][r];
                size_t idx = (size_t)(mb + r) * ldc + n;
                if constexpr (EPI == 0)      ((float*)Cv)[idx] = v;
                else if constexpr (EPI == 1) ((u16*)Cv)[idx]   = f2bf(v);
                else if constexpr (EPI == 2) ((float*)Cv)[idx] = v + bias[n];
                else if constexpr (EPI == 3) ((u16*)Cv)[idx]   = f2bf(v + bias[n]);
                else if constexpr (EPI == 4) ((float*)Cv)[idx] = v * alpha * keyvec[n];
                else if constexpr (EPI == 5) ((float*)Cv)[idx] = v * alpha;
                else if constexpr (EPI == 6) ((u8*)Cv)[idx]    = (v * alpha > beta) ? (u8)1 : (u8)0;
            }
        }
    }
}

// ---------------------------------------------------------------------------
// f32 -> bf16 elementwise
// ---------------------------------------------------------------------------
__global__ void cvt_f32_bf16(const float* __restrict__ src, u16* __restrict__ dst, int n) {
    int i = blockIdx.x * 256 + threadIdx.x;
    if (i < n) dst[i] = f2bf(src[i]);
}

// ---------------------------------------------------------------------------
// Split qkv[N,768] (f32) into per-head normalized q,k and raw/normalized v,
// all stored as bf16 [N, H*D] (head-major columns).
// grid (N, 12): y -> {mat 0..2} x {head 0..3}; block 64 (=D)
// ---------------------------------------------------------------------------
__global__ void normalize_qkv(const float* __restrict__ qkv,
                              u16* __restrict__ qn, u16* __restrict__ kn,
                              u16* __restrict__ vraw, u16* __restrict__ vn)
{
    __shared__ float red[64];
    int n = blockIdx.x, g = blockIdx.y;
    int mat = g >> 2, h = g & 3, d = threadIdx.x;
    float v = qkv[(size_t)n * 768 + mat * 256 + h * 64 + d];
    red[d] = v * v;
    __syncthreads();
    for (int s = 32; s > 0; s >>= 1) {
        if (d < s) red[d] += red[d + s];
        __syncthreads();
    }
    float inv = 1.0f / fmaxf(sqrtf(red[0]), 1e-6f);
    size_t oi = (size_t)n * 256 + h * 64 + d;
    if (mat == 0)      qn[oi] = f2bf(v * inv);
    else if (mat == 1) kn[oi] = f2bf(v * inv);
    else { vraw[oi] = f2bf(v); vn[oi] = f2bf(v * inv); }
}

// ---------------------------------------------------------------------------
// block reductions (256 threads)
// ---------------------------------------------------------------------------
__device__ __forceinline__ float blockReduceMax(float v, float* sdata) {
    int t = threadIdx.x;
    sdata[t] = v; __syncthreads();
    for (int s = 128; s > 0; s >>= 1) {
        if (t < s) sdata[t] = fmaxf(sdata[t], sdata[t + s]);
        __syncthreads();
    }
    float r = sdata[0]; __syncthreads();
    return r;
}
__device__ __forceinline__ float blockReduceSum(float v, float* sdata) {
    int t = threadIdx.x;
    sdata[t] = v; __syncthreads();
    for (int s = 128; s > 0; s >>= 1) {
        if (t < s) sdata[t] += sdata[t + s];
        __syncthreads();
    }
    float r = sdata[0]; __syncthreads();
    return r;
}

// ---------------------------------------------------------------------------
// Fused dual softmax + combine for one head:
//   attn = 0.5*(softmax(lc) + softmax(lr)) (bf16), sim += attn/H (f32).
// grid = NROW rows, block 256, 12 elements/thread in registers.
// ---------------------------------------------------------------------------
__global__ __launch_bounds__(256) void softmax_combine(
    const float* __restrict__ lc, const float* __restrict__ lr,
    u16* __restrict__ attn, float* __restrict__ sim, int h)
{
    __shared__ float sdata[256];
    int row = blockIdx.x, t = threadIdx.x;
    const float* pc = lc + (size_t)row * NROW;
    const float* pr = lr + (size_t)row * NROW;
    float xc[12], xr[12];
    float mC = -1e30f, mR = -1e30f;
    #pragma unroll
    for (int j = 0; j < 12; j++) {
        int i = t + j * 256;
        xc[j] = pc[i]; xr[j] = pr[i];
        mC = fmaxf(mC, xc[j]); mR = fmaxf(mR, xr[j]);
    }
    mC = blockReduceMax(mC, sdata);
    mR = blockReduceMax(mR, sdata);
    float sC = 0.f, sR = 0.f;
    #pragma unroll
    for (int j = 0; j < 12; j++) {
        sC += __expf(xc[j] - mC);
        sR += __expf(xr[j] - mR);
    }
    sC = blockReduceSum(sC, sdata);
    sR = blockReduceSum(sR, sdata);
    float rC = 1.0f / sC, rR = 1.0f / sR;
    u16*  pa = attn + (size_t)row * NROW;
    float* ps = sim + (size_t)row * NROW;
    #pragma unroll
    for (int j = 0; j < 12; j++) {
        int i = t + j * 256;
        float a = 0.5f * (__expf(xc[j] - mC) * rC + __expf(xr[j] - mR) * rR);
        pa[i] = f2bf(a);
        float s = (h == 0 ? 0.0f : ps[i]) + 0.25f * a;
        ps[i] = s;
    }
}

// ---------------------------------------------------------------------------
// sim_round2 / obj_mask from sim_attn + gram masks. softmax denom cancels:
//   sim_round2 = mc*e^(x-M) / sum(mc*e^(x-M));  obj = mr*sr2 / sum(mc*sr2)
// ---------------------------------------------------------------------------
__global__ __launch_bounds__(256) void maskpool(
    const float* __restrict__ sim, const u8* __restrict__ mcm,
    const u8* __restrict__ mrm, u16* __restrict__ sr2, u16* __restrict__ obj)
{
    __shared__ float sdata[256];
    int row = blockIdx.x, t = threadIdx.x;
    const float* ps = sim + (size_t)row * NROW;
    const u8* pc = mcm + (size_t)row * NROW;
    const u8* pr = mrm + (size_t)row * NROW;
    float x[12], mc[12];
    float mx = -1e30f;
    #pragma unroll
    for (int j = 0; j < 12; j++) {
        int i = t + j * 256;
        x[j] = ps[i]; mc[j] = (float)pc[i];
        mx = fmaxf(mx, x[j]);
    }
    mx = blockReduceMax(mx, sdata);
    float e[12];
    float sme = 0.f;
    #pragma unroll
    for (int j = 0; j < 12; j++) {
        e[j] = __expf(x[j] - mx);
        sme += mc[j] * e[j];
    }
    sme = blockReduceSum(sme, sdata);
    float inv = 1.0f / sme;
    float v[12];
    float s3 = 0.f;
    u16* po = sr2 + (size_t)row * NROW;
    #pragma unroll
    for (int j = 0; j < 12; j++) {
        int i = t + j * 256;
        v[j] = mc[j] * e[j] * inv;
        po[i] = f2bf(v[j]);
        s3 += mc[j] * v[j];
    }
    s3 = blockReduceSum(s3, sdata);
    float inv3 = 1.0f / s3;
    u16* pq = obj + (size_t)row * NROW;
    #pragma unroll
    for (int j = 0; j < 12; j++) {
        int i = t + j * 256;
        float o = (float)pr[i] * v[j] * inv3;
        pq[i] = f2bf(o);
    }
}

// ---------------------------------------------------------------------------
// strided bf16 copy (for concat halves)
// ---------------------------------------------------------------------------
__global__ void copy_bf16(const u16* __restrict__ src, int sld,
                          u16* __restrict__ dst, int dld, int rows, int cols)
{
    int idx = blockIdx.x * 256 + threadIdx.x;
    int total = rows * cols;
    if (idx < total) {
        int r = idx / cols, c = idx % cols;
        dst[(size_t)r * dld + c] = src[(size_t)r * sld + c];
    }
}

// ---------------------------------------------------------------------------
extern "C" void kernel_launch(void* const* d_in, const int* in_sizes, int n_in,
                              void* d_out, int out_size, void* d_ws, size_t ws_size,
                              hipStream_t stream)
{
    (void)in_sizes; (void)n_in; (void)out_size; (void)ws_size;
    const int N = NROW;

    const float* cls_features = (const float*)d_in[0];   // [N,256]
    const float* reg_features = (const float*)d_in[1];   // [N,64]
    const float* cls_scores   = (const float*)d_in[2];   // [N]
    const float* reg_proj_w   = (const float*)d_in[3];   // [256,64]
    const float* qkv_cls_w    = (const float*)d_in[4];   // [768,256]
    const float* qkv_reg_w    = (const float*)d_in[5];   // [768,256]
    const float* l1_cls_w     = (const float*)d_in[6];   // [512,512]
    const float* l1_cls_b     = (const float*)d_in[7];
    const float* l1_reg_w     = (const float*)d_in[8];
    const float* l1_reg_b     = (const float*)d_in[9];
    const float* l2_cls_w     = (const float*)d_in[10];  // [1024,1024]
    const float* l2_cls_b     = (const float*)d_in[11];
    const float* l2_reg_w     = (const float*)d_in[12];
    const float* l2_reg_b     = (const float*)d_in[13];

    // ---- workspace carve (256B aligned), total ~255 MB ----
    char* base = (char*)d_ws;
    size_t off = 0;
    auto alloc = [&](size_t bytes) -> void* {
        void* p = base + off;
        off += (bytes + 255) & ~(size_t)255;
        return p;
    };
    u16* cls_bf   = (u16*)alloc((size_t)N * 256 * 2);
    u16* reg_bf   = (u16*)alloc((size_t)N * 64 * 2);
    u16* wproj    = (u16*)alloc(256 * 64 * 2);
    u16* wqkv_c   = (u16*)alloc(768 * 256 * 2);
    u16* wqkv_r   = (u16*)alloc(768 * 256 * 2);
    u16* wl1_c    = (u16*)alloc(512 * 512 * 2);
    u16* wl1_r    = (u16*)alloc(512 * 512 * 2);
    u16* wl2_c    = (u16*)alloc(1024 * 1024 * 2);
    u16* wl2_r    = (u16*)alloc(1024 * 1024 * 2);
    u16* reg_com  = (u16*)alloc((size_t)N * 256 * 2);
    float* qkv_cf = (float*)alloc((size_t)N * 768 * 4);
    float* qkv_rf = (float*)alloc((size_t)N * 768 * 4);
    u16* qcn = (u16*)alloc((size_t)N * 256 * 2);
    u16* kcn = (u16*)alloc((size_t)N * 256 * 2);
    u16* vc  = (u16*)alloc((size_t)N * 256 * 2);
    u16* vcn = (u16*)alloc((size_t)N * 256 * 2);
    u16* qrn = (u16*)alloc((size_t)N * 256 * 2);
    u16* krn = (u16*)alloc((size_t)N * 256 * 2);
    u16* vr  = (u16*)alloc((size_t)N * 256 * 2);
    u16* vrn = (u16*)alloc((size_t)N * 256 * 2);
    float* log_c = (float*)alloc((size_t)N * N * 4);
    float* log_r = (float*)alloc((size_t)N * N * 4);
    u16* attn_h  = (u16*)alloc((size_t)N * N * 2);   // one head at a time
    float* sim   = (float*)alloc((size_t)N * N * 4);
    u8* mask_c   = (u8*)alloc((size_t)N * N);
    u8* mask_r   = (u8*)alloc((size_t)N * N);
    u16* sr2     = (u16*)alloc((size_t)N * N * 2);
    u16* obj     = (u16*)alloc((size_t)N * N * 2);
    u16* enh_c   = (u16*)alloc((size_t)N * 512 * 2);
    u16* enh_r   = (u16*)alloc((size_t)N * 512 * 2);
    u16* t_c     = (u16*)alloc((size_t)N * 512 * 2);
    u16* t_r     = (u16*)alloc((size_t)N * 512 * 2);
    u16* pool_c  = (u16*)alloc((size_t)N * 1024 * 2);
    u16* pool_r  = (u16*)alloc((size_t)N * 1024 * 2);

    float* out_cls = (float*)d_out;
    float* out_reg = out_cls + (size_t)N * 1024;

    auto cvt = [&](const float* s, u16* d, int n) {
        cvt_f32_bf16<<<(n + 255) / 256, 256, 0, stream>>>(s, d, n);
    };
    // ---- convert activations + weights to bf16 ----
    cvt(cls_features, cls_bf, N * 256);
    cvt(reg_features, reg_bf, N * 64);
    cvt(reg_proj_w, wproj, 256 * 64);
    cvt(qkv_cls_w, wqkv_c, 768 * 256);
    cvt(qkv_reg_w, wqkv_r, 768 * 256);
    cvt(l1_cls_w, wl1_c, 512 * 512);
    cvt(l1_reg_w, wl1_r, 512 * 512);
    cvt(l2_cls_w, wl2_c, 1024 * 1024);
    cvt(l2_reg_w, wl2_r, 1024 * 1024);

    auto grid = [](int M, int Nn) { return dim3((Nn + 127) / 128, (M + 127) / 128); };

    // ---- reg projection: [N,64] x [256,64]^T -> bf16 [N,256] ----
    gemm_bf16<true, 1><<<grid(N, 256), 256, 0, stream>>>(
        reg_bf, 64, wproj, 64, reg_com, 256, N, 256, 64, nullptr, 0.f, 0.f, nullptr);

    // ---- qkv: [N,256] x [768,256]^T -> f32 [N,768] ----
    gemm_bf16<true, 0><<<grid(N, 768), 256, 0, stream>>>(
        cls_bf, 256, wqkv_c, 256, qkv_cf, 768, N, 768, 256, nullptr, 0.f, 0.f, nullptr);
    gemm_bf16<true, 0><<<grid(N, 768), 256, 0, stream>>>(
        reg_com, 256, wqkv_r, 256, qkv_rf, 768, N, 768, 256, nullptr, 0.f, 0.f, nullptr);

    // ---- per-head L2-normalize q,k; v raw + normalized ----
    normalize_qkv<<<dim3(N, 12), 64, 0, stream>>>(qkv_cf, qcn, kcn, vc, vcn);
    normalize_qkv<<<dim3(N, 12), 64, 0, stream>>>(qkv_rf, qrn, krn, vr, vrn);

    // ---- gram similarity masks: (Vn Vn^T / H > thresh) ----
    gemm_bf16<true, 6><<<grid(N, N), 256, 0, stream>>>(
        vcn, 256, vcn, 256, mask_c, N, N, N, 256, nullptr, 0.25f, 0.75f, nullptr);
    gemm_bf16<true, 6><<<grid(N, N), 256, 0, stream>>>(
        vrn, 256, vrn, 256, mask_r, N, N, N, 256, nullptr, 0.25f, 0.99f, nullptr);

    // ---- per-head attention: logits -> dual softmax combine -> attn @ V ----
    for (int h = 0; h < NHEAD; h++) {
        const int co = h * HD;
        gemm_bf16<true, 4><<<grid(N, N), 256, 0, stream>>>(      // 25*score[key]*qk
            qcn + co, 256, kcn + co, 256, log_c, N, N, N, HD,
            nullptr, 25.0f, 0.f, cls_scores);
        gemm_bf16<true, 5><<<grid(N, N), 256, 0, stream>>>(      // 25*qk
            qrn + co, 256, krn + co, 256, log_r, N, N, N, HD,
            nullptr, 25.0f, 0.f, nullptr);
        softmax_combine<<<N, 256, 0, stream>>>(log_c, log_r, attn_h, sim, h);
        // inter[h] = attn @ v[h]  -> enh[:, h*64 : h*64+64]
        gemm_bf16<false, 1><<<grid(N, HD), 256, 0, stream>>>(
            attn_h, N, vc + co, 256, enh_c + co, 512, N, HD, N,
            nullptr, 0.f, 0.f, nullptr);
        gemm_bf16<false, 1><<<grid(N, HD), 256, 0, stream>>>(
            attn_h, N, vr + co, 256, enh_r + co, 512, N, HD, N,
            nullptr, 0.f, 0.f, nullptr);
    }
    // ---- enh = concat(inter, ori) : copy v into cols 256..511 ----
    copy_bf16<<<(N * 256 + 255) / 256, 256, 0, stream>>>(vc, 256, enh_c + 256, 512, N, 256);
    copy_bf16<<<(N * 256 + 255) / 256, 256, 0, stream>>>(vr, 256, enh_r + 256, 512, N, 256);

    // ---- sim_round2 / obj_mask ----
    maskpool<<<N, 256, 0, stream>>>(sim, mask_c, mask_r, sr2, obj);

    // ---- l1: t = enh @ W1^T + b1 (bf16) ----
    gemm_bf16<true, 3><<<grid(N, 512), 256, 0, stream>>>(
        enh_c, 512, wl1_c, 512, t_c, 512, N, 512, 512, l1_cls_b, 0.f, 0.f, nullptr);
    gemm_bf16<true, 3><<<grid(N, 512), 256, 0, stream>>>(
        enh_r, 512, wl1_r, 512, t_r, 512, N, 512, 512, l1_reg_b, 0.f, 0.f, nullptr);

    // ---- pooled = concat(mask @ t, t) ----
    gemm_bf16<false, 1><<<grid(N, 512), 256, 0, stream>>>(
        sr2, N, t_c, 512, pool_c, 1024, N, 512, N, nullptr, 0.f, 0.f, nullptr);
    gemm_bf16<false, 1><<<grid(N, 512), 256, 0, stream>>>(
        obj, N, t_r, 512, pool_r, 1024, N, 512, N, nullptr, 0.f, 0.f, nullptr);
    copy_bf16<<<(N * 512 + 255) / 256, 256, 0, stream>>>(t_c, 512, pool_c + 512, 1024, N, 512);
    copy_bf16<<<(N * 512 + 255) / 256, 256, 0, stream>>>(t_r, 512, pool_r + 512, 1024, N, 512);

    // ---- l2: out = pooled @ W2^T + b2 (f32 -> d_out) ----
    gemm_bf16<true, 2><<<grid(N, 1024), 256, 0, stream>>>(
        pool_c, 1024, wl2_c, 1024, out_cls, 1024, N, 1024, 1024, l2_cls_b, 0.f, 0.f, nullptr);
    gemm_bf16<true, 2><<<grid(N, 1024), 256, 0, stream>>>(
        pool_r, 1024, wl2_r, 1024, out_reg, 1024, N, 1024, 1024, l2_reg_b, 0.f, 0.f, nullptr);
}
